// CrossAttentionFusionGated_54571854463182
// MI455X (gfx1250) — compile-verified, matrix operand order checked by reading, not run on hardware
//
#include <hip/hip_runtime.h>
#include <hip/hip_bf16.h>
#include <math.h>

// ---- problem constants ----
constexpr int kB   = 8;
constexpr int kT   = 4096;
constexpr int kD   = 512;
constexpr int kH   = 8;
constexpr int kDH  = 64;
constexpr int kSPK = 4;
constexpr int kHS  = kH * kSPK;   // 32 folded "keys" per batch
constexpr int kHG  = 128;
constexpr int TILE = 32;          // token rows per block
constexpr float kEPS = 1e-5f;

typedef __attribute__((ext_vector_type(16))) _Float16 v16h;
typedef __attribute__((ext_vector_type(8)))  float    v8f;
typedef __attribute__((ext_vector_type(4)))  float    f4v;
typedef __attribute__((ext_vector_type(4)))  _Float16 h4v;
typedef __attribute__((ext_vector_type(8)))  _Float16 h8v;

// Gather one 16x32 WMMA fragment (A or B) from a row-major f16 matrix.
// p points at [row_of_this_lane][k0 + kb]; lanes 0-15 take K {0..7,16..23},
// lanes 16-31 take K {8..15,24..31} (kb handles the 8-offset).
__device__ __forceinline__ v16h frag_h(const _Float16* p) {
  h8v lo = *(const h8v*)p;
  h8v hi = *(const h8v*)(p + 16);
  v16h f;
  #pragma unroll
  for (int j = 0; j < 8; ++j) { f[j] = lo[j]; f[8 + j] = hi[j]; }
  return f;
}

// ============================================================
// Precompute 1: spk_tokens = relu(speaker_emb @ We^T + be) + spk_pos
// ============================================================
__global__ void spk_kernel(const float* __restrict__ se, const float* __restrict__ We,
                           const float* __restrict__ be, const float* __restrict__ spos,
                           float* __restrict__ spk) {
  int idx = blockIdx.x * blockDim.x + threadIdx.x;
  if (idx >= kB * kSPK * kD) return;
  int b = idx / (kSPK * kD);
  int r = idx - b * (kSPK * kD);           // r = s*D + d
  const float* x = se + (size_t)b * kD;
  const float* w = We + (size_t)r * kD;
  float acc = be[r];
  for (int j = 0; j < kD; ++j) acc = fmaf(x[j], w[j], acc);
  spk[idx] = fmaxf(acc, 0.f) + spos[r];
}

// ============================================================
// Precompute 2: k = spk@Wk^T + bk ; v = spk@Wv^T + bv
// ============================================================
__global__ void kv_kernel(const float* __restrict__ spk,
                          const float* __restrict__ Wk, const float* __restrict__ bk,
                          const float* __restrict__ Wv, const float* __restrict__ bv,
                          float* __restrict__ kbuf, float* __restrict__ vbuf) {
  int idx = blockIdx.x * blockDim.x + threadIdx.x;
  const int n = kB * kSPK * kD;
  if (idx >= 2 * n) return;
  int which = idx >= n;
  int r = which ? idx - n : idx;
  int bs = r / kD, d = r - bs * kD;
  const float* x = spk + (size_t)bs * kD;
  const float* W = (which ? Wv : Wk) + (size_t)d * kD;
  float acc = which ? bv[d] : bk[d];
  for (int j = 0; j < kD; ++j) acc = fmaf(x[j], W[j], acc);
  (which ? vbuf : kbuf)[r] = acc;
}

// ============================================================
// Precompute 3: fold Wq into K -> M (f16) + Msum, fold Wo into V -> VO^T (f16),
//               score bias bq.k, gamma-folded Wg1 (f16) + beta-folded bg1.
// ============================================================
__global__ void fold_kernel(const float* __restrict__ kbuf, const float* __restrict__ vbuf,
                            const float* __restrict__ Wq, const float* __restrict__ bq,
                            const float* __restrict__ Wo, const float* __restrict__ Wg1,
                            const float* __restrict__ bg1,
                            const float* __restrict__ glng, const float* __restrict__ glnb,
                            _Float16* __restrict__ Mh, _Float16* __restrict__ VOt,
                            float* __restrict__ cb, float* __restrict__ Msum,
                            _Float16* __restrict__ Wg1h, float* __restrict__ bg1p) {
  int idx = blockIdx.x * blockDim.x + threadIdx.x;
  const int nM  = kB * kHS * kD;
  const int nV  = kB * kD * kHS;
  const int nC  = kB * kHS;
  const int nS  = kB * kHS;
  const int nG  = kHG * kD;
  const int nGB = kHG;
  if (idx < nM) {
    // M[b][hs][d] = (1/8) * sum_dh Wq[h*64+dh][d] * k[b][s][h*64+dh]
    int b = idx / (kHS * kD); int rem = idx - b * (kHS * kD);
    int hs = rem / kD; int d = rem - hs * kD;
    int h = hs >> 2, s = hs & 3;
    const float* kr = kbuf + (size_t)(b * kSPK + s) * kD + h * kDH;
    float acc = 0.f;
    for (int j = 0; j < kDH; ++j)
      acc = fmaf(Wq[(size_t)(h * kDH + j) * kD + d], kr[j], acc);
    Mh[idx] = (_Float16)(acc * 0.125f);
  } else if (idx < nM + nV) {
    // VOt[b][d][hs] = sum_dh Wo[d][h*64+dh] * v[b][s][h*64+dh]
    int i = idx - nM;
    int b = i / (kD * kHS); int rem = i - b * (kD * kHS);
    int d = rem / kHS; int hs = rem - d * kHS;
    int h = hs >> 2, s = hs & 3;
    const float* vr = vbuf + (size_t)(b * kSPK + s) * kD + h * kDH;
    const float* wo = Wo + (size_t)d * kD + h * kDH;
    float acc = 0.f;
    for (int j = 0; j < kDH; ++j) acc = fmaf(wo[j], vr[j], acc);
    VOt[i] = (_Float16)acc;
  } else if (idx < nM + nV + nC) {
    // cb[b][hs] = (1/8) * bq . k_head
    int i = idx - nM - nV; int b = i / kHS; int hs = i - b * kHS;
    int h = hs >> 2, s = hs & 3;
    const float* kr = kbuf + (size_t)(b * kSPK + s) * kD + h * kDH;
    float acc = 0.f;
    for (int j = 0; j < kDH; ++j) acc = fmaf(bq[h * kDH + j], kr[j], acc);
    cb[i] = acc * 0.125f;
  } else if (idx < nM + nV + nC + nS) {
    // Msum[b][hs] = sum_d M[b][hs][d] (fp32, recomputed exactly)
    int i = idx - nM - nV - nC; int b = i / kHS; int hs = i - b * kHS;
    int h = hs >> 2, s = hs & 3;
    const float* kr = kbuf + (size_t)(b * kSPK + s) * kD + h * kDH;
    float acc = 0.f;
    for (int j = 0; j < kDH; ++j) {
      const float* wr = Wq + (size_t)(h * kDH + j) * kD;
      float rsum = 0.f;
      for (int d = 0; d < kD; ++d) rsum += wr[d];
      acc = fmaf(kr[j], rsum, acc);
    }
    Msum[i] = acc * 0.125f;
  } else if (idx < nM + nV + nC + nS + nG) {
    // Wg1h = Wg1 * gamma (gamma of gate-LN folded into weights)
    int i = idx - nM - nV - nC - nS;
    int k = i & (kD - 1);
    Wg1h[i] = (_Float16)(Wg1[i] * glng[k]);
  } else if (idx < nM + nV + nC + nS + nG + nGB) {
    // bg1p = bg1 + Wg1 . beta (beta of gate-LN folded into bias)
    int n = idx - nM - nV - nC - nS - nG;
    const float* wr = Wg1 + (size_t)n * kD;
    float acc = bg1[n];
    for (int k = 0; k < kD; ++k) acc = fmaf(wr[k], glnb[k], acc);
    bg1p[n] = acc;
  }
}

// ============================================================
// Fused main kernel: gate GEMM + scores + softmax + combine + final LN.
// LDS holds the pre-affine normalized tile x' = (x-mu)*rstd (f16);
// gamma/beta are folded into Wg1h/bg1p, and raw-x scores are recovered via
// scores = irs * (x'.M^T) + mu*Msum + cb.
// ============================================================
__global__ __launch_bounds__(256) void fused_kernel(
    const float* __restrict__ X,
    const _Float16* __restrict__ Mh,    // (B, HS, D)
    const _Float16* __restrict__ VOt,   // (B, D, HS)
    const float* __restrict__ cb,       // (B, HS)
    const float* __restrict__ Msum,     // (B, HS)
    const _Float16* __restrict__ Wg1h,  // (HG, D), gamma-folded
    const float* __restrict__ bg1p,     // (HG), beta-folded
    const float* __restrict__ Wg2, const float* __restrict__ bg2,
    const float* __restrict__ bo,
    const float* __restrict__ lng, const float* __restrict__ lnb,
    float* __restrict__ out) {
  __shared__ _Float16 sXh[TILE * kD];    // x' (f16); later reused for attn (f16)
  __shared__ _Float16 sGh[TILE * kHG];
  __shared__ float sSc[TILE * kHS];
  __shared__ float sW[TILE * kHS];
  __shared__ float sPs[TILE * 8];
  __shared__ float sPq[TILE * 8];
  __shared__ float sMu[TILE], sRs[TILE], sIrs[TILE], sG[TILE];

  const int t = threadIdx.x;
  const int lane = t & 31;
  const int wave = t >> 5;
  const int r15 = lane & 15;
  const int hi = lane >> 4;          // 0 or 1 (wave32 half)
  const int kb = hi * 8;             // fragment K sub-offset per ISA layout

  const size_t rowg0 = (size_t)blockIdx.x * TILE;
  const int b = (int)(rowg0 >> 12);  // tile never crosses batch (4096 % 32 == 0)

  const size_t AW_OFF = (size_t)kB * kT * kD;
  const size_t G_OFF  = AW_OFF + (size_t)kB * kT * kSPK;

  const int row = t >> 3, ch = t & 7;
  const f4v* xrow = (const f4v*)(X + (rowg0 + row) * kD + ch * 64);

  // ---- Phase 0a: LN stats (vector loads, 16B) ----
  {
    float s = 0.f, q = 0.f;
    #pragma unroll
    for (int j = 0; j < 16; ++j) {
      f4v v = xrow[j];
      #pragma unroll
      for (int e = 0; e < 4; ++e) { s += v[e]; q += v[e] * v[e]; }
    }
    sPs[row * 8 + ch] = s;
    sPq[row * 8 + ch] = q;
  }
  __syncthreads();
  if (t < TILE) {
    float s = 0.f, q = 0.f;
    #pragma unroll
    for (int c = 0; c < 8; ++c) { s += sPs[t * 8 + c]; q += sPq[t * 8 + c]; }
    float mu = s * (1.f / kD);
    float var = q * (1.f / kD) - mu * mu;
    sMu[t]  = mu;
    sRs[t]  = rsqrtf(var + kEPS);
    sIrs[t] = sqrtf(var + kEPS);
  }
  __syncthreads();

  // ---- Phase 0b: write x' = (x-mu)*rstd to LDS (f16, b64 stores) ----
  {
    const float mu = sMu[row], rs = sRs[row];
    h4v* op = (h4v*)&sXh[row * kD + ch * 64];
    #pragma unroll
    for (int j = 0; j < 16; ++j) {
      f4v v = xrow[j];
      h4v h;
      #pragma unroll
      for (int e = 0; e < 4; ++e) h[e] = (_Float16)((v[e] - mu) * rs);
      op[j] = h;
    }
  }
  __syncthreads();

  // ---- Gate GEMM: relu(x' @ Wg1h^T + bg1p)  (M=32,N=128,K=512) ----
  #pragma unroll
  for (int tt = 0; tt < 2; ++tt) {
    const int tid = wave + tt * 8;
    const int mt = tid >> 3, nt = tid & 7;
    const int arow = mt * 16 + r15;
    v8f acc = {};
    for (int k0 = 0; k0 < kD; k0 += 32) {
      v16h a  = frag_h(&sXh[arow * kD + k0 + kb]);
      v16h bm = frag_h(Wg1h + (size_t)(nt * 16 + r15) * kD + k0 + kb);
      acc = __builtin_amdgcn_wmma_f32_16x16x32_f16(false, a, false, bm,
                                                   (short)0, acc, false, false);
    }
    const int col = nt * 16 + r15;
    const float bb = bg1p[col];
    #pragma unroll
    for (int r = 0; r < 8; ++r)
      sGh[(mt * 16 + hi * 8 + r) * kHG + col] = (_Float16)fmaxf(acc[r] + bb, 0.f);
  }
  __syncthreads();

  // ---- Gate head on wave 4 (concurrent with scores on waves 0-3) ----
  if (wave == 4) {
    const int rw = lane;
    float g = bg2[0];
    const h8v* gp = (const h8v*)&sGh[rw * kHG];
    #pragma unroll 4
    for (int j = 0; j < 16; ++j) {
      h8v h = gp[j];
      #pragma unroll
      for (int e = 0; e < 8; ++e) g = fmaf((float)h[e], Wg2[j * 8 + e], g);
    }
    g = 1.f / (1.f + expf(-g));
    sG[rw] = g;
    out[G_OFF + rowg0 + rw] = g;
  }

  // ---- Scores GEMM: irs*(x' @ M^T) + mu*Msum + cb  (M=32,N=32,K=512) ----
  if (wave < 4) {
    const int mt = wave >> 1, nt = wave & 1;
    const int arow = mt * 16 + r15;
    v8f acc = {};
    for (int k0 = 0; k0 < kD; k0 += 32) {
      v16h a  = frag_h(&sXh[arow * kD + k0 + kb]);
      v16h bm = frag_h(Mh + ((size_t)b * kHS + nt * 16 + r15) * kD + k0 + kb);
      acc = __builtin_amdgcn_wmma_f32_16x16x32_f16(false, a, false, bm,
                                                   (short)0, acc, false, false);
    }
    const int col = nt * 16 + r15;
    const float c0 = cb[b * kHS + col];
    const float ms = Msum[b * kHS + col];
    #pragma unroll
    for (int r = 0; r < 8; ++r) {
      const int rr = mt * 16 + hi * 8 + r;
      sSc[rr * kHS + col] = fmaf(acc[r], sIrs[rr], fmaf(sMu[rr], ms, c0));
    }
  }
  __syncthreads();

  // ---- Softmax over SPK=4 within each head (one thread per row-head) ----
  {
    const int rw = t >> 3, h = t & 7;
    const float* sp = &sSc[rw * kHS + h * 4];
    float s0 = sp[0], s1 = sp[1], s2 = sp[2], s3 = sp[3];
    float m = fmaxf(fmaxf(s0, s1), fmaxf(s2, s3));
    float e0 = expf(s0 - m), e1 = expf(s1 - m), e2 = expf(s2 - m), e3 = expf(s3 - m);
    float inv = 1.f / (e0 + e1 + e2 + e3);
    float* wp = &sW[rw * kHS + h * 4];
    wp[0] = e0 * inv; wp[1] = e1 * inv; wp[2] = e2 * inv; wp[3] = e3 * inv;
  }
  __syncthreads();
  if (t < TILE * kSPK) {  // head-averaged attention weights
    int rw = t >> 2, s = t & 3;
    float aw = 0.f;
    #pragma unroll
    for (int h = 0; h < kH; ++h) aw += sW[rw * kHS + h * 4 + s];
    out[AW_OFF + (rowg0 + rw) * kSPK + s] = aw * (1.f / kH);
  }

  // ---- Combine GEMM: attn = w @ VO  (M=32,N=512,K=32) -> overwrite sXh ----
  #pragma unroll
  for (int i = 0; i < 8; ++i) {
    const int tid = wave * 8 + i;
    const int mt = tid >> 5, nt = tid & 31;
    v16h a, bm;
    {
      const f4v* wp = (const f4v*)&sW[(mt * 16 + r15) * kHS + kb];
      f4v w0 = wp[0], w1 = wp[1];   // K = kb .. kb+7
      f4v w2 = wp[4], w3 = wp[5];   // K = kb+16 .. kb+23
      #pragma unroll
      for (int e = 0; e < 4; ++e) {
        a[e]      = (_Float16)w0[e];
        a[4 + e]  = (_Float16)w1[e];
        a[8 + e]  = (_Float16)w2[e];
        a[12 + e] = (_Float16)w3[e];
      }
    }
    bm = frag_h(VOt + ((size_t)b * kD + nt * 16 + r15) * kHS + kb);
    v8f acc = {};
    acc = __builtin_amdgcn_wmma_f32_16x16x32_f16(false, a, false, bm,
                                                 (short)0, acc, false, false);
    const int col = nt * 16 + r15;
    #pragma unroll
    for (int r = 0; r < 8; ++r)
      sXh[(mt * 16 + hi * 8 + r) * kD + col] = (_Float16)acc[r];
  }
  __syncthreads();

  // ---- Epilogue: fused = LN(x + g*(attn + bo)) ----
  {
    const float g = sG[row];
    const h4v* av = (const h4v*)&sXh[row * kD + ch * 64];
    const f4v* bov = (const f4v*)(bo + ch * 64);
    float s = 0.f, q = 0.f;
    #pragma unroll
    for (int j = 0; j < 16; ++j) {
      f4v v = xrow[j]; h4v a = av[j]; f4v bb = bov[j];
      #pragma unroll
      for (int e = 0; e < 4; ++e) {
        float f = v[e] + g * ((float)a[e] + bb[e]);
        s += f; q += f * f;
      }
    }
    sPs[row * 8 + ch] = s;
    sPq[row * 8 + ch] = q;
  }
  __syncthreads();
  if (t < TILE) {
    float s = 0.f, q = 0.f;
    #pragma unroll
    for (int c = 0; c < 8; ++c) { s += sPs[t * 8 + c]; q += sPq[t * 8 + c]; }
    float mu = s * (1.f / kD);
    float var = q * (1.f / kD) - mu * mu;
    sMu[t] = mu;
    sRs[t] = rsqrtf(var + kEPS);
  }
  __syncthreads();
  {
    const float g = sG[row];
    const float mu = sMu[row], rs = sRs[row];
    const h4v* av = (const h4v*)&sXh[row * kD + ch * 64];
    const f4v* bov = (const f4v*)(bo + ch * 64);
    const f4v* lgv = (const f4v*)(lng + ch * 64);
    const f4v* lbv = (const f4v*)(lnb + ch * 64);
    f4v* op = (f4v*)(out + (rowg0 + row) * kD + ch * 64);
    #pragma unroll
    for (int j = 0; j < 16; ++j) {
      f4v v = xrow[j]; h4v a = av[j]; f4v bb = bov[j];
      f4v lg = lgv[j], lb = lbv[j], o;
      #pragma unroll
      for (int e = 0; e < 4; ++e) {
        float f = v[e] + g * ((float)a[e] + bb[e]);
        o[e] = (f - mu) * rs * lg[e] + lb[e];
      }
      op[j] = o;
    }
  }
}

// ============================================================
extern "C" void kernel_launch(void* const* d_in, const int* in_sizes, int n_in,
                              void* d_out, int out_size, void* d_ws, size_t ws_size,
                              hipStream_t stream) {
  (void)in_sizes; (void)n_in; (void)out_size; (void)ws_size;
  const float* token = (const float*)d_in[0];
  const float* se    = (const float*)d_in[1];
  const float* We    = (const float*)d_in[2];
  const float* be    = (const float*)d_in[3];
  const float* spos  = (const float*)d_in[4];
  const float* Wq    = (const float*)d_in[5];
  const float* bq    = (const float*)d_in[6];
  const float* Wk    = (const float*)d_in[7];
  const float* bk    = (const float*)d_in[8];
  const float* Wv    = (const float*)d_in[9];
  const float* bv    = (const float*)d_in[10];
  const float* Wo    = (const float*)d_in[11];
  const float* bo    = (const float*)d_in[12];
  const float* glng  = (const float*)d_in[13];
  const float* glnb  = (const float*)d_in[14];
  const float* Wg1   = (const float*)d_in[15];
  const float* bg1   = (const float*)d_in[16];
  const float* Wg2   = (const float*)d_in[17];
  const float* bg2   = (const float*)d_in[18];
  const float* lng   = (const float*)d_in[19];
  const float* lnb   = (const float*)d_in[20];
  float* out = (float*)d_out;

  char* ws = (char*)d_ws;
  float*    spk  = (float*)(ws + 0);            // 64 KB
  float*    kbuf = (float*)(ws + 65536);        // 64 KB
  float*    vbuf = (float*)(ws + 131072);       // 64 KB
  _Float16* Mh   = (_Float16*)(ws + 196608);    // 256 KB
  _Float16* VOt  = (_Float16*)(ws + 458752);    // 256 KB
  float*    cbf  = (float*)(ws + 720896);       // 1 KB
  float*    Msum = (float*)(ws + 721920);       // 1 KB
  float*    bg1p = (float*)(ws + 722944);       // 512 B
  _Float16* Wg1h = (_Float16*)(ws + 723456);    // 128 KB

  spk_kernel<<<(kB * kSPK * kD + 255) / 256, 256, 0, stream>>>(se, We, be, spos, spk);
  kv_kernel<<<(2 * kB * kSPK * kD + 255) / 256, 256, 0, stream>>>(spk, Wk, bk, Wv, bv, kbuf, vbuf);
  const int nfold = kB * kHS * kD + kB * kD * kHS + kB * kHS + kB * kHS + kHG * kD + kHG;
  fold_kernel<<<(nfold + 255) / 256, 256, 0, stream>>>(kbuf, vbuf, Wq, bq, Wo, Wg1, bg1,
                                                       glng, glnb, Mh, VOt, cbf, Msum, Wg1h, bg1p);
  fused_kernel<<<(kB * kT) / TILE, 256, 0, stream>>>(token, Mh, VOt, cbf, Msum, Wg1h, bg1p,
                                                     Wg2, bg2, bo, lng, lnb, out);
}
